// Block_51367808860482
// MI455X (gfx1250) — compile-verified, matrix operand order checked by reading, not run on hardware
//
#include <hip/hip_runtime.h>
#include <hip/hip_bf16.h>
#include <math.h>

// ---------------------------------------------------------------------------
// Types for CDNA5 WMMA (wave32): v_wmma_f32_16x16x32_bf16
// ---------------------------------------------------------------------------
typedef __attribute__((ext_vector_type(16))) __bf16 bf16x16;
typedef __attribute__((ext_vector_type(8)))  __bf16 bf16x8;
typedef __attribute__((ext_vector_type(8)))  float  f32x8;

#define NPTS 65536
#define CDIM 256
#define NHEAD 8
#define HDIM 32
#define WIN 128

__device__ __forceinline__ __bf16 f2bf(float x) { return (__bf16)x; }

__device__ __forceinline__ f32x8 zf8() {
  f32x8 z;
#pragma unroll
  for (int i = 0; i < 8; ++i) z[i] = 0.f;
  return z;
}

__device__ __forceinline__ f32x8 wmma_bf16(bf16x16 a, bf16x16 b, f32x8 c) {
  return __builtin_amdgcn_wmma_f32_16x16x32_bf16(
      /*neg_a=*/false, a, /*neg_b=*/false, b,
      /*c_mod=*/(short)0, c, /*reuse_a=*/false, /*reuse_b=*/false);
}

// A-fragment (16x32 bf16) loader from a row-major tile (stride ld elems).
// Layout (ISA 7.12.2): lane L: M=L&15, half=L>>4; VGPR0-3 hold K=half*8+0..7,
// VGPR4-7 hold K=16+half*8+0..7  -> two b128 loads per lane.
// B-fragments for out = x @ W^T (W row-major [out][in]) use the same
// addressing with the N index in lane&15.
__device__ __forceinline__ bf16x16 frag_ld(const __bf16* tile, int ld, int k0) {
  int lane = threadIdx.x & 31;
  int r = lane & 15, hv = lane >> 4;
  const __bf16* p = tile + (size_t)r * ld + k0 + hv * 8;
  bf16x8 lo = *(const bf16x8*)p;
  bf16x8 hi = *(const bf16x8*)(p + 16);
  return __builtin_shufflevector(lo, hi, 0,1,2,3,4,5,6,7,8,9,10,11,12,13,14,15);
}

// Per-lane row pointer variant (each lane already points at its own row):
// unconditional loads -- invalid rows must point at a zero row.
__device__ __forceinline__ bf16x16 frag_ld_rp(const __bf16* rowp, int k0) {
  int lane = threadIdx.x & 31;
  int hv = lane >> 4;
  const __bf16* p = rowp + k0 + hv * 8;
  bf16x8 lo = *(const bf16x8*)p;
  bf16x8 hi = *(const bf16x8*)(p + 16);
  return __builtin_shufflevector(lo, hi, 0,1,2,3,4,5,6,7,8,9,10,11,12,13,14,15);
}

__device__ __forceinline__ float wave_sum32(float v) {
#pragma unroll
  for (int m = 1; m < 32; m <<= 1) v += __shfl_xor(v, m, 32);
  return v;
}

// ---------------------------------------------------------------------------
// Weight / activation casts + zero pad row
// ---------------------------------------------------------------------------
__global__ void k_cast(const float* __restrict__ in, __bf16* __restrict__ out, int n) {
  for (int i = blockIdx.x * 256 + threadIdx.x; i < n; i += gridDim.x * 256)
    out[i] = f2bf(in[i]);
}

__global__ void k_zero_bf16(__bf16* __restrict__ p, int n) {
  int i = blockIdx.x * 256 + threadIdx.x;
  if (i < n) p[i] = (__bf16)0.f;
}

// cpe_w [27][c][d] f32 -> wt [27][d][c] bf16 (so B-tiles load like A-tiles)
__global__ void k_cast_cpeT(const float* __restrict__ w, __bf16* __restrict__ wt) {
  const int total = 27 * 256 * 256;
  for (int i = blockIdx.x * 256 + threadIdx.x; i < total; i += gridDim.x * 256) {
    int k = i >> 16, r = i & 65535, d = r >> 8, c = r & 255;
    wt[i] = f2bf(w[((size_t)k << 16) | (c << 8) | d]);
  }
}

// ---------------------------------------------------------------------------
// CPE gathered GEMM: cpe1[n,d] = sum_k gather_k(feat)[n,:] @ cpe_wT[k] + cpe_b
// block = 256 thr = 8 waves; tile 64 rows x 32 cols; K = 27*256.
// featb has NPTS+1 rows; row NPTS is all zeros (invalid neighbors map there),
// so every fragment load is unconditional (no EXEC masking in the hot loop).
// ---------------------------------------------------------------------------
__global__ __launch_bounds__(256)
void k_cpe_gather(const __bf16* __restrict__ featb, const __bf16* __restrict__ cpeWt,
                  const float* __restrict__ cpe_b, const int* __restrict__ nbr,
                  __bf16* __restrict__ cpe1) {
  int wave = threadIdx.x >> 5, lane = threadIdx.x & 31;
  int r16 = lane & 15, hv = lane >> 4;
  int m0 = blockIdx.y * 64 + (wave >> 1) * 16;
  int n0 = blockIdx.x * 32 + (wave & 1) * 16;
  int myrow = m0 + r16;

  // Preload all 27 neighbor indices (clamped to the zero row) up front so the
  // address chain is off the critical path of the MAC loop.
  int nb[27];
#pragma unroll
  for (int k = 0; k < 27; ++k) {
    int idx = nbr[(size_t)myrow * 27 + k];
    nb[k] = (idx < 0) ? NPTS : idx;
  }

  f32x8 acc0 = zf8(), acc1 = zf8();
  for (int k = 0; k < 27; ++k) {
    const __bf16* rowp = featb + (size_t)nb[k] * CDIM;
    const __bf16* Wrow = cpeWt + ((size_t)k * CDIM + n0) * CDIM;
#pragma unroll
    for (int c0 = 0; c0 < CDIM; c0 += 64) {  // dual accumulators: 2 indep chains
      acc0 = wmma_bf16(frag_ld_rp(rowp, c0),      frag_ld(Wrow, CDIM, c0),      acc0);
      acc1 = wmma_bf16(frag_ld_rp(rowp, c0 + 32), frag_ld(Wrow, CDIM, c0 + 32), acc1);
    }
  }
  int col = n0 + r16;
  float bv = cpe_b[col];
#pragma unroll
  for (int e = 0; e < 8; ++e) {
    int row = m0 + e + 8 * hv;
    cpe1[(size_t)row * CDIM + col] = f2bf(acc0[e] + acc1[e] + bv);
  }
}

// ---------------------------------------------------------------------------
// Generic bf16 GEMM: out = A @ Wt^T + bias (+res). No LDS; fragments direct.
// ---------------------------------------------------------------------------
#define EPI_F32 0
#define EPI_RESF32 1

template <int EPI, int KD>
__global__ __launch_bounds__(256)
void k_gemm(const __bf16* __restrict__ A, const __bf16* __restrict__ Wt,
            const float* __restrict__ bias, const float* __restrict__ res,
            float* __restrict__ outF, int Nout) {
  int wave = threadIdx.x >> 5, lane = threadIdx.x & 31;
  int r16 = lane & 15, hv = lane >> 4;
  int m0 = blockIdx.y * 64 + (wave >> 1) * 16;
  int n0 = blockIdx.x * 32 + (wave & 1) * 16;
  const __bf16* Arow = A + (size_t)m0 * KD;
  const __bf16* Wrow = Wt + (size_t)n0 * KD;
  f32x8 acc0 = zf8(), acc1 = zf8();
#pragma unroll
  for (int k0 = 0; k0 < KD; k0 += 64) {
    acc0 = wmma_bf16(frag_ld(Arow, KD, k0),      frag_ld(Wrow, KD, k0),      acc0);
    acc1 = wmma_bf16(frag_ld(Arow, KD, k0 + 32), frag_ld(Wrow, KD, k0 + 32), acc1);
  }
  int col = n0 + r16;
  float bv = bias[col];
#pragma unroll
  for (int e = 0; e < 8; ++e) {
    int row = m0 + e + 8 * hv;
    size_t idx = (size_t)row * Nout + col;
    float v = acc0[e] + acc1[e] + bv;
    if (EPI == EPI_RESF32) v += res[idx];
    outF[idx] = v;
  }
}

// ---------------------------------------------------------------------------
// LayerNorm kernels: one wave per row (C=256 = 32 lanes x 8)
// ---------------------------------------------------------------------------
__global__ __launch_bounds__(256)
void k_fuse_ln1(const float* __restrict__ cpe2, const float* __restrict__ feat,
                const float* __restrict__ g0, const float* __restrict__ b0,
                const float* __restrict__ g1, const float* __restrict__ b1,
                float* __restrict__ feat1, __bf16* __restrict__ xln1) {
  int wave = threadIdx.x >> 5, lane = threadIdx.x & 31;
  size_t row = (size_t)blockIdx.x * 8 + wave;
  const float* xr = cpe2 + row * CDIM;
  const float* fr = feat + row * CDIM;
  float x[8], y[8];
  float s = 0.f;
#pragma unroll
  for (int i = 0; i < 8; ++i) { x[i] = xr[lane + 32 * i]; s += x[i]; }
  float mean = wave_sum32(s) * (1.f / 256.f);
  float vs = 0.f;
#pragma unroll
  for (int i = 0; i < 8; ++i) { float d = x[i] - mean; vs += d * d; }
  float rstd = rsqrtf(wave_sum32(vs) * (1.f / 256.f) + 1e-5f);
  float s2 = 0.f;
#pragma unroll
  for (int i = 0; i < 8; ++i) {
    int c = lane + 32 * i;
    y[i] = (x[i] - mean) * rstd * g0[c] + b0[c] + fr[c];
    feat1[row * CDIM + c] = y[i];
    s2 += y[i];
  }
  float mean2 = wave_sum32(s2) * (1.f / 256.f);
  float vs2 = 0.f;
#pragma unroll
  for (int i = 0; i < 8; ++i) { float d = y[i] - mean2; vs2 += d * d; }
  float rstd2 = rsqrtf(wave_sum32(vs2) * (1.f / 256.f) + 1e-5f);
#pragma unroll
  for (int i = 0; i < 8; ++i) {
    int c = lane + 32 * i;
    xln1[row * CDIM + c] = f2bf((y[i] - mean2) * rstd2 * g1[c] + b1[c]);
  }
}

__global__ __launch_bounds__(256)
void k_ln(const float* __restrict__ x, const float* __restrict__ g,
          const float* __restrict__ b, __bf16* __restrict__ out) {
  int wave = threadIdx.x >> 5, lane = threadIdx.x & 31;
  size_t row = (size_t)blockIdx.x * 8 + wave;
  const float* xr = x + row * CDIM;
  float v[8];
  float s = 0.f;
#pragma unroll
  for (int i = 0; i < 8; ++i) { v[i] = xr[lane + 32 * i]; s += v[i]; }
  float mean = wave_sum32(s) * (1.f / 256.f);
  float vs = 0.f;
#pragma unroll
  for (int i = 0; i < 8; ++i) { float d = v[i] - mean; vs += d * d; }
  float rstd = rsqrtf(wave_sum32(vs) * (1.f / 256.f) + 1e-5f);
#pragma unroll
  for (int i = 0; i < 8; ++i) {
    int c = lane + 32 * i;
    out[row * CDIM + c] = f2bf((v[i] - mean) * rstd * g[c] + b[c]);
  }
}

// ---------------------------------------------------------------------------
// Fused QKV + windowed attention. Block = (window p, head h), 256 thr = 8 waves,
// wave w owns rows w*16..w*16+15 of the 128-row window.
// Dynamic LDS: sQ[128*32] sK[128*32] sVt[32*128] sAttn[128*128] bf16 = 56 KB
// ---------------------------------------------------------------------------
__global__ __launch_bounds__(256)
void k_attn(const __bf16* __restrict__ xln1, const __bf16* __restrict__ qkvWt,
            const float* __restrict__ qkv_b, const int* __restrict__ order,
            __bf16* __restrict__ attnout) {
  extern __shared__ __align__(16) char smem_raw[];
  __bf16* sQ = (__bf16*)smem_raw;        // [128][32]
  __bf16* sK = sQ + 128 * 32;            // [128][32]
  __bf16* sVt = sK + 128 * 32;           // [32][128]  (v transposed)
  __bf16* sAttn = sVt + 32 * 128;        // [128][128]

  int p = blockIdx.x, h = blockIdx.y;
  int wave = threadIdx.x >> 5, lane = threadIdx.x & 31;
  int r16 = lane & 15, hv = lane >> 4;

  // ---- step 1: q,k,v for this wave's 16 rows (gathered through `order`)
  int pos = p * WIN + wave * 16 + r16;
  const __bf16* Arow = xln1 + (size_t)order[pos] * CDIM;
  f32x8 accq[2], acck[2], accv[2];
#pragma unroll
  for (int t = 0; t < 2; ++t) { accq[t] = zf8(); acck[t] = zf8(); accv[t] = zf8(); }
#pragma unroll
  for (int k0 = 0; k0 < CDIM; k0 += 32) {
    bf16x16 a = frag_ld_rp(Arow, k0);
#pragma unroll
    for (int t = 0; t < 2; ++t) {
      bf16x16 bq = frag_ld(qkvWt + (size_t)(0 * CDIM + h * HDIM + t * 16) * CDIM, CDIM, k0);
      bf16x16 bk = frag_ld(qkvWt + (size_t)(1 * CDIM + h * HDIM + t * 16) * CDIM, CDIM, k0);
      bf16x16 bv = frag_ld(qkvWt + (size_t)(2 * CDIM + h * HDIM + t * 16) * CDIM, CDIM, k0);
      accq[t] = wmma_bf16(a, bq, accq[t]);
      acck[t] = wmma_bf16(a, bk, acck[t]);
      accv[t] = wmma_bf16(a, bv, accv[t]);
    }
  }
  const float scale = 0.17677669529663687f;  // (C/H)^-0.5 = 32^-0.5
#pragma unroll
  for (int t = 0; t < 2; ++t) {
    float bq = qkv_b[0 * CDIM + h * HDIM + t * 16 + r16];
    float bk = qkv_b[1 * CDIM + h * HDIM + t * 16 + r16];
    float bv = qkv_b[2 * CDIM + h * HDIM + t * 16 + r16];
#pragma unroll
    for (int e = 0; e < 8; ++e) {
      int row = wave * 16 + e + 8 * hv;   // 0..127
      int d = t * 16 + r16;               // 0..31
      sQ[row * 32 + d] = f2bf((accq[t][e] + bq) * scale);
      sK[row * 32 + d] = f2bf(acck[t][e] + bk);
      sVt[d * 128 + row] = f2bf(accv[t][e] + bv);
    }
  }
  __syncthreads();

  // ---- step 2: scores = q @ k^T  (16 rows x 128 cols per wave, single K=32)
  f32x8 sc[8];
  {
    bf16x16 aq = frag_ld(sQ + wave * 16 * 32, 32, 0);
#pragma unroll
    for (int ct = 0; ct < 8; ++ct)
      sc[ct] = wmma_bf16(aq, frag_ld(sK + ct * 16 * 32, 32, 0), zf8());
  }
  // ---- softmax over 128 cols; rows live in a 16-lane half per f32 C-layout
#pragma unroll
  for (int e = 0; e < 8; ++e) {
    float m = -3.0e30f;
#pragma unroll
    for (int ct = 0; ct < 8; ++ct) m = fmaxf(m, sc[ct][e]);
#pragma unroll
    for (int d = 1; d < 16; d <<= 1) m = fmaxf(m, __shfl_xor(m, d, 32));
    float pv[8], s = 0.f;
#pragma unroll
    for (int ct = 0; ct < 8; ++ct) { pv[ct] = __expf(sc[ct][e] - m); s += pv[ct]; }
#pragma unroll
    for (int d = 1; d < 16; d <<= 1) s += __shfl_xor(s, d, 32);
    float inv = 1.f / s;
    int row = wave * 16 + e + 8 * hv;
#pragma unroll
    for (int ct = 0; ct < 8; ++ct)
      sAttn[row * 128 + ct * 16 + r16] = f2bf(pv[ct] * inv);
  }
  __syncthreads();

  // ---- step 3: out = attn @ v  (16 x 32 per wave, K=128)
  f32x8 o[2] = {zf8(), zf8()};
#pragma unroll
  for (int k0 = 0; k0 < 128; k0 += 32) {
    bf16x16 a = frag_ld(sAttn + wave * 16 * 128, 128, k0);
#pragma unroll
    for (int t = 0; t < 2; ++t)
      o[t] = wmma_bf16(a, frag_ld(sVt + t * 16 * 128, 128, k0), o[t]);
  }
#pragma unroll
  for (int t = 0; t < 2; ++t)
#pragma unroll
    for (int e = 0; e < 8; ++e) {
      int row = wave * 16 + e + 8 * hv;
      int node = order[p * WIN + row];  // folds reference's [inverse] scatter
      attnout[(size_t)node * CDIM + h * HDIM + t * 16 + r16] = f2bf(o[t][e]);
    }
}

// ---------------------------------------------------------------------------
// Fused MLP: out = feat2 + fc2( gelu( fc1(xln2) ) ). Block = 64 rows.
// Hidden (64 x 1024) kept in LDS in two 512-col phases (64 KB dynamic LDS).
// ---------------------------------------------------------------------------
__global__ __launch_bounds__(256)
void k_mlp(const __bf16* __restrict__ xln2, const __bf16* __restrict__ fc1Wt,
           const float* __restrict__ fc1_b, const __bf16* __restrict__ fc2Wt,
           const float* __restrict__ fc2_b, const float* __restrict__ feat2,
           float* __restrict__ out) {
  extern __shared__ __align__(16) char smem_raw[];
  __bf16* sH = (__bf16*)smem_raw;  // [64][512]

  int wave = threadIdx.x >> 5, lane = threadIdx.x & 31;
  int r16 = lane & 15, hv = lane >> 4;
  int mt = wave >> 1;                       // 0..3 row-tile in block
  int m0 = blockIdx.x * 64 + mt * 16;

  // Hoist A fragments (16 x 256) for this wave's rows
  const __bf16* Arow = xln2 + (size_t)m0 * CDIM;
  bf16x16 afr[8];
#pragma unroll
  for (int i = 0; i < 8; ++i) afr[i] = frag_ld(Arow, CDIM, i * 32);

  f32x8 oacc[8];
#pragma unroll
  for (int t = 0; t < 8; ++t) oacc[t] = zf8();

  for (int phase = 0; phase < 2; ++phase) {
    int cbase = phase * 512;
    // fc1 + GELU for hidden cols [cbase, cbase+512)
#pragma unroll 4
    for (int u = 0; u < 16; ++u) {
      int ct = (wave & 1) + 2 * u;          // 0..31 col tile within phase
      int colg = cbase + ct * 16;
      const __bf16* Wrow = fc1Wt + (size_t)colg * CDIM;
      f32x8 a0 = zf8(), a1 = zf8();
#pragma unroll
      for (int i = 0; i < 8; i += 2) {       // dual accumulators
        a0 = wmma_bf16(afr[i],     frag_ld(Wrow, CDIM, i * 32),       a0);
        a1 = wmma_bf16(afr[i + 1], frag_ld(Wrow, CDIM, (i + 1) * 32), a1);
      }
      float bv = fc1_b[colg + r16];
#pragma unroll
      for (int e = 0; e < 8; ++e) {
        float xx = a0[e] + a1[e] + bv;
        float g = 0.5f * xx * (1.f + erff(xx * 0.70710678118654752f));  // exact gelu
        int row = mt * 16 + e + 8 * hv;
        sH[row * 512 + ct * 16 + r16] = f2bf(g);
      }
    }
    __syncthreads();
    // fc2 partial accumulation over this phase's 512 K values
#pragma unroll
    for (int t = 0; t < 8; ++t) {
      int ct = (wave & 1) + 2 * t;          // out col tile 0..15
      const __bf16* Wrow = fc2Wt + (size_t)(ct * 16) * 1024 + cbase;
      const __bf16* Ash = sH + (size_t)(mt * 16) * 512;
#pragma unroll
      for (int k0 = 0; k0 < 512; k0 += 32)
        oacc[t] = wmma_bf16(frag_ld(Ash, 512, k0), frag_ld(Wrow, 1024, k0), oacc[t]);
    }
    __syncthreads();
  }
  // epilogue: + fc2_b + feat2 residual -> f32 output
#pragma unroll
  for (int t = 0; t < 8; ++t) {
    int col = ((wave & 1) + 2 * t) * 16 + r16;
    float bv = fc2_b[col];
#pragma unroll
    for (int e = 0; e < 8; ++e) {
      int row = m0 + e + 8 * hv;
      size_t idx = (size_t)row * CDIM + col;
      out[idx] = oacc[t][e] + bv + feat2[idx];
    }
  }
}

// ---------------------------------------------------------------------------
// Host launcher
// ---------------------------------------------------------------------------
extern "C" void kernel_launch(void* const* d_in, const int* in_sizes, int n_in,
                              void* d_out, int out_size, void* d_ws, size_t ws_size,
                              hipStream_t stream) {
  const float* feat      = (const float*)d_in[0];
  const float* cpe_w     = (const float*)d_in[1];
  const float* cpe_b     = (const float*)d_in[2];
  const float* cpe_lin_w = (const float*)d_in[3];
  const float* cpe_lin_b = (const float*)d_in[4];
  const float* cpe_ln_g  = (const float*)d_in[5];
  const float* cpe_ln_b  = (const float*)d_in[6];
  const float* ln1_g     = (const float*)d_in[7];
  const float* ln1_b     = (const float*)d_in[8];
  const float* qkv_w     = (const float*)d_in[9];
  const float* qkv_b     = (const float*)d_in[10];
  const float* proj_w    = (const float*)d_in[11];
  const float* proj_b    = (const float*)d_in[12];
  const float* ln2_g     = (const float*)d_in[13];
  const float* ln2_b     = (const float*)d_in[14];
  const float* fc1_w     = (const float*)d_in[15];
  const float* fc1_b     = (const float*)d_in[16];
  const float* fc2_w     = (const float*)d_in[17];
  const float* fc2_b     = (const float*)d_in[18];
  const int*   nbr_idx   = (const int*)d_in[19];
  const int*   order     = (const int*)d_in[20];
  float*       outF      = (float*)d_out;

  char* ws = (char*)d_ws;
  size_t off = 0;
  auto alloc = [&](size_t bytes) -> char* {
    char* p = ws + off;
    off += (bytes + 255) & ~(size_t)255;
    return p;
  };
  __bf16* wb_cpeT   = (__bf16*)alloc((size_t)27 * 256 * 256 * 2);
  __bf16* wb_cpelin = (__bf16*)alloc((size_t)256 * 256 * 2);
  __bf16* wb_qkv    = (__bf16*)alloc((size_t)768 * 256 * 2);
  __bf16* wb_proj   = (__bf16*)alloc((size_t)256 * 256 * 2);
  __bf16* wb_fc1    = (__bf16*)alloc((size_t)1024 * 256 * 2);
  __bf16* wb_fc2    = (__bf16*)alloc((size_t)256 * 1024 * 2);
  // bufA: featbf (NPTS+1 rows, last row zero) -> xln1 -> xln2
  __bf16* bufA      = (__bf16*)alloc((size_t)(NPTS + 1) * CDIM * 2);
  __bf16* bufB      = (__bf16*)alloc((size_t)NPTS * CDIM * 2);  // cpe1 -> attnout
  float*  bufC      = (float*)alloc((size_t)NPTS * CDIM * 4);   // cpe2 -> feat2
  float*  feat1     = (float*)alloc((size_t)NPTS * CDIM * 4);
  if (off > ws_size) return;  // workspace too small; bail rather than corrupt

  // 1) casts (weights bf16; cpe_w transposed), feat -> bf16 (+ zero row)
  k_cast<<<4096, 256, 0, stream>>>(feat, bufA, NPTS * CDIM);
  k_zero_bf16<<<1, 256, 0, stream>>>(bufA + (size_t)NPTS * CDIM, CDIM);
  k_cast<<<256, 256, 0, stream>>>(cpe_lin_w, wb_cpelin, 256 * 256);
  k_cast<<<768, 256, 0, stream>>>(qkv_w, wb_qkv, 768 * 256);
  k_cast<<<256, 256, 0, stream>>>(proj_w, wb_proj, 256 * 256);
  k_cast<<<1024, 256, 0, stream>>>(fc1_w, wb_fc1, 1024 * 256);
  k_cast<<<1024, 256, 0, stream>>>(fc2_w, wb_fc2, 256 * 1024);
  k_cast_cpeT<<<4096, 256, 0, stream>>>(cpe_w, wb_cpeT);

  // 2) CPE gathered GEMM (K = 27*256) -> cpe1 (bf16)
  k_cpe_gather<<<dim3(CDIM / 32, NPTS / 64), 256, 0, stream>>>(
      bufA, wb_cpeT, cpe_b, nbr_idx, bufB);

  // 3) cpe_lin GEMM -> cpe2 (f32)
  k_gemm<EPI_F32, CDIM><<<dim3(CDIM / 32, NPTS / 64), 256, 0, stream>>>(
      bufB, wb_cpelin, cpe_lin_b, nullptr, bufC, CDIM);

  // 4) feat1 = feat + LN(cpe2); xln1 = LN(feat1) (bf16)
  k_fuse_ln1<<<NPTS / 8, 256, 0, stream>>>(
      bufC, feat, cpe_ln_g, cpe_ln_b, ln1_g, ln1_b, feat1, bufA);

  // 5) fused qkv + windowed attention -> attnout (bf16, scattered by order)
  k_attn<<<dim3(NPTS / WIN, NHEAD), 256, 57344, stream>>>(
      bufA, wb_qkv, qkv_b, order, bufB);

  // 6) proj GEMM + residual -> feat2 (f32)
  k_gemm<EPI_RESF32, CDIM><<<dim3(CDIM / 32, NPTS / 64), 256, 0, stream>>>(
      bufB, wb_proj, proj_b, feat1, bufC, CDIM);

  // 7) xln2 = LN(feat2) (bf16)
  k_ln<<<NPTS / 8, 256, 0, stream>>>(bufC, ln2_g, ln2_b, bufA);

  // 8) fused MLP (fc1 + exact GELU + fc2) + residual -> d_out (f32)
  k_mlp<<<NPTS / 64, 256, 65536, stream>>>(
      bufA, wb_fc1, fc1_b, wb_fc2, fc2_b, bufC, outF);
}
